// BitSelfAttention_3040836846382
// MI455X (gfx1250) — matrix operand order repaired and checked by reading, not compiled, not run on hardware
//
#include <hip/hip_runtime.h>

// ---------------------------------------------------------------------------
// BitLinear self-attention for gfx1250 (MI455X), wave32 + v_wmma_f32_16x16x32_f16
// ---------------------------------------------------------------------------
typedef _Float16 v16h __attribute__((ext_vector_type(16)));
typedef _Float16 v8h  __attribute__((ext_vector_type(8)));
typedef _Float16 v4h  __attribute__((ext_vector_type(4)));
typedef float    v8f  __attribute__((ext_vector_type(8)));

#define D_MODEL 2048
#define T_SEQ   2048
#define N_HEAD  16
#define D_HEAD  128
#define B_SZ    4
#define M_ROWS  (B_SZ * T_SEQ)        // 8192
#define W_ELEMS (D_MODEL * D_MODEL)   // 4194304

__device__ __forceinline__ v8f wmma32(v16h a, v16h b, v8f c) {
  // D = A(16x32 f16) x B(32x16 f16) + C(16x16 f32)
  return __builtin_amdgcn_wmma_f32_16x16x32_f16(false, a, false, b, (short)0, c,
                                                false, false);
}

// A-fragment (16x32 f16): lane m=lane%16; hf=lane/16 selects contiguous K half (k = 16*hf + s)
__device__ __forceinline__ v16h load_a16(const _Float16* row, int k, int hf) {
  union { v16h v; v8h h[2]; } u;
  u.h[0] = *(const v8h*)(row + k + hf * 16);
  u.h[1] = *(const v8h*)(row + k + hf * 16 + 8);
  return u.v;
}

// ---------------------------------------------------------------------------
// gamma = max(mean(|w|), 1e-5): deterministic two-phase reduction (f4 loads)
// ---------------------------------------------------------------------------
__global__ __launch_bounds__(256) void abs_sum_partial(const float4* __restrict__ w,
                                                       float* __restrict__ part,
                                                       int n4) {
  __shared__ float sm[256];
  float s = 0.f;
  for (int i = blockIdx.x * 256 + threadIdx.x; i < n4; i += 256 * gridDim.x) {
    float4 v = w[i];
    s += fabsf(v.x) + fabsf(v.y) + fabsf(v.z) + fabsf(v.w);
  }
  sm[threadIdx.x] = s;
  __syncthreads();
  for (int st = 128; st > 0; st >>= 1) {
    if ((int)threadIdx.x < st) sm[threadIdx.x] += sm[threadIdx.x + st];
    __syncthreads();
  }
  if (threadIdx.x == 0) part[blockIdx.x] = sm[0];
}

__global__ __launch_bounds__(256) void gamma_final(const float* __restrict__ part,
                                                   int np, float invN,
                                                   float* __restrict__ gOut) {
  __shared__ float sm[256];
  float s = 0.f;
  for (int i = threadIdx.x; i < np; i += 256) s += part[i];
  sm[threadIdx.x] = s;
  __syncthreads();
  for (int st = 128; st > 0; st >>= 1) {
    if ((int)threadIdx.x < st) sm[threadIdx.x] += sm[threadIdx.x + st];
    __syncthreads();
  }
  if (threadIdx.x == 0) gOut[0] = fmaxf(sm[0] * invN, 1e-5f);
}

// w_q = clip(round(w/gamma), -1, 1) as f16; 4 elems/thread
__global__ __launch_bounds__(256) void quant_w(const float4* __restrict__ w,
                                               const float* __restrict__ g,
                                               v4h* __restrict__ out, int n4) {
  int i = blockIdx.x * 256 + threadIdx.x;
  if (i < n4) {
    const float inv = 1.f / g[0];
    float4 v = w[i];
    v4h o;
    o[0] = (_Float16)fminf(1.f, fmaxf(-1.f, rintf(v.x * inv)));
    o[1] = (_Float16)fminf(1.f, fmaxf(-1.f, rintf(v.y * inv)));
    o[2] = (_Float16)fminf(1.f, fmaxf(-1.f, rintf(v.z * inv)));
    o[3] = (_Float16)fminf(1.f, fmaxf(-1.f, rintf(v.w * inv)));
    out[i] = o;
  }
}

__global__ __launch_bounds__(256) void cvt_f16(const float4* __restrict__ x,
                                               v4h* __restrict__ out, int n4) {
  int i = blockIdx.x * 256 + threadIdx.x;
  if (i < n4) {
    float4 v = x[i];
    v4h o;
    o[0] = (_Float16)v.x; o[1] = (_Float16)v.y;
    o[2] = (_Float16)v.z; o[3] = (_Float16)v.w;
    out[i] = o;
  }
}

// ---------------------------------------------------------------------------
// BitLinear GEMM: Out[m,n] = gamma*extra * sum_k A[m,k]*W[n,k]
// block = 128 threads (4 waves, 2x2), wave tile 64x64 (4x4 WMMA frags)
// MODE 0: f16 row-major   MODE 1: f32 row-major   MODE 2: f16 Vt[b,h,d,t]
// ---------------------------------------------------------------------------
template <int MODE>
__global__ __launch_bounds__(128) void gemm_bit(const _Float16* __restrict__ A,
                                                const _Float16* __restrict__ W,
                                                const float* __restrict__ gammaPtr,
                                                void* __restrict__ Out,
                                                float extraScale) {
  constexpr int K = D_MODEL, N = D_MODEL;
  const int lane = threadIdx.x & 31, wave = threadIdx.x >> 5;
  const int ln = lane & 15, hf = lane >> 4;
  const int wM = blockIdx.y * 128 + (wave >> 1) * 64;
  const int wN = blockIdx.x * 128 + (wave & 1) * 64;

  v8f acc[4][4] = {};
  const _Float16* aRow[4];
  const _Float16* bRow[4];
#pragma unroll
  for (int i = 0; i < 4; ++i) {
    aRow[i] = A + (size_t)(wM + i * 16 + ln) * K;
    bRow[i] = W + (size_t)(wN + i * 16 + ln) * K;
  }

  for (int k = 0; k < K; k += 32) {
    v16h aF[4], bF[4];
#pragma unroll
    for (int mt = 0; mt < 4; ++mt) aF[mt] = load_a16(aRow[mt], k, hf);
#pragma unroll
    for (int nt = 0; nt < 4; ++nt)
      bF[nt] = *(const v16h*)(bRow[nt] + k + hf * 16);   // 32B contiguous
#pragma unroll
    for (int mt = 0; mt < 4; ++mt)
#pragma unroll
      for (int nt = 0; nt < 4; ++nt)
        acc[mt][nt] = wmma32(aF[mt], bF[nt], acc[mt][nt]);
  }

  const float g = gammaPtr[0] * extraScale;

  if constexpr (MODE == 0) {
    _Float16* O = (_Float16*)Out;
#pragma unroll
    for (int mt = 0; mt < 4; ++mt)
#pragma unroll
      for (int nt = 0; nt < 4; ++nt) {
        const int col = wN + nt * 16 + ln;
#pragma unroll
        for (int r = 0; r < 8; ++r) {
          const int row = wM + mt * 16 + 8 * hf + r;
          O[(size_t)row * N + col] = (_Float16)(acc[mt][nt][r] * g);
        }
      }
  } else if constexpr (MODE == 1) {
    float* O = (float*)Out;
#pragma unroll
    for (int mt = 0; mt < 4; ++mt)
#pragma unroll
      for (int nt = 0; nt < 4; ++nt) {
        const int col = wN + nt * 16 + ln;
#pragma unroll
        for (int r = 0; r < 8; ++r) {
          const int row = wM + mt * 16 + 8 * hf + r;
          O[(size_t)row * N + col] = acc[mt][nt][r] * g;
        }
      }
  } else {  // MODE 2: Vt[((b*16+h)*128+d)*T + t] — contiguous 16B store per tile
    _Float16* O = (_Float16*)Out;
#pragma unroll
    for (int mt = 0; mt < 4; ++mt) {
      const int row0 = wM + mt * 16 + 8 * hf;   // flat b*T + t, multiple of 8
      const int b = row0 >> 11, t0 = row0 & (T_SEQ - 1);
#pragma unroll
      for (int nt = 0; nt < 4; ++nt) {
        const int col = wN + nt * 16 + ln;      // h*128 + d
        const int hIdx = col >> 7, dIdx = col & 127;
        v8h pk;
#pragma unroll
        for (int r = 0; r < 8; ++r) pk[r] = (_Float16)(acc[mt][nt][r] * g);
        _Float16* dst =
            O + ((size_t)((b * N_HEAD + hIdx) * D_HEAD + dIdx)) * T_SEQ + t0;
        *(v8h*)dst = pk;
      }
    }
  }
}

// ---------------------------------------------------------------------------
// Flash attention step: S = Q K^T, online softmax, O += P V.
// MASKED=true only for the single diagonal key-block per wave.
// Wave-private LDS tile for C-layout -> A-layout P transpose; no block
// barriers needed (per-wave LDS ops are in-order; explicit s_wait_dscnt).
// ---------------------------------------------------------------------------
template <bool MASKED>
__device__ __forceinline__ void flash_step(
    int ks, int qRow0, int ln, int hf,
    const v16h (&qF)[4], const _Float16* __restrict__ kBase,
    const _Float16* __restrict__ vBase, _Float16* __restrict__ pW,
    v8f (&oAcc)[8], float (&mS)[8], float (&lS)[8]) {
  // ---- S = Q K^T (16x32) ----
  v8f sAcc[2] = {};
#pragma unroll
  for (int c = 0; c < 4; ++c) {
#pragma unroll
    for (int j = 0; j < 2; ++j) {
      v16h bF = *(const v16h*)(kBase + (size_t)(ks + j * 16 + ln) * D_MODEL +
                               c * 32 + hf * 16);
      sAcc[j] = wmma32(qF[c], bF, sAcc[j]);
    }
  }
  // prefetch next key/value block into caches (speculative, gfx1250 path)
  __builtin_prefetch(kBase + (size_t)(ks + 32 + ln) * D_MODEL + hf * 16, 0, 1);
  __builtin_prefetch(vBase + (size_t)ln * T_SEQ + ks + 32 + hf * 16, 0, 1);

  // ---- (mask +) online softmax; row stats within 16-lane halves ----
  float p0v[8], p1v[8];
#pragma unroll
  for (int r = 0; r < 8; ++r) {
    float s0 = sAcc[0][r];
    float s1 = sAcc[1][r];
    if (MASKED) {
      const int qAbs = qRow0 + r + 8 * hf;
      if (ks + ln      > qAbs) s0 = -1e30f;
      if (ks + 16 + ln > qAbs) s1 = -1e30f;
    }
    float mx = fmaxf(s0, s1);
#pragma unroll
    for (int m = 1; m < 16; m <<= 1) mx = fmaxf(mx, __shfl_xor(mx, m, 32));
    const float mNew = fmaxf(mS[r], mx);
    const float alpha = __expf(mS[r] - mNew);
    const float p0 = __expf(s0 - mNew);
    const float p1 = __expf(s1 - mNew);
    float rs = p0 + p1;
#pragma unroll
    for (int m = 1; m < 16; m <<= 1) rs += __shfl_xor(rs, m, 32);
    mS[r] = mNew;
    lS[r] = lS[r] * alpha + rs;
    p0v[r] = p0; p1v[r] = p1;
#pragma unroll
    for (int nt = 0; nt < 8; ++nt) oAcc[nt][r] *= alpha;
  }

  // ---- P: C-layout -> A-layout via wave-private LDS tile ----
#pragma unroll
  for (int r = 0; r < 8; ++r) {
    pW[(r + 8 * hf) * 32 + ln]      = (_Float16)p0v[r];
    pW[(r + 8 * hf) * 32 + 16 + ln] = (_Float16)p1v[r];
  }
  asm volatile("s_wait_dscnt 0x0" ::: "memory");  // wave-internal LDS fence
  union { v16h v; v8h hh[2]; } pa;
  pa.hh[0] = *(const v8h*)&pW[ln * 32 + hf * 16];
  pa.hh[1] = *(const v8h*)&pW[ln * 32 + hf * 16 + 8];
  asm volatile("" ::: "memory");  // keep reads before next iteration's writes

  // ---- O += P V  (B operand from Vt: contiguous 32B per lane) ----
#pragma unroll
  for (int nt = 0; nt < 8; ++nt) {
    v16h bV = *(const v16h*)(vBase + (size_t)(nt * 16 + ln) * T_SEQ + ks +
                             hf * 16);
    oAcc[nt] = wmma32(pa.v, bV, oAcc[nt]);
  }
}

// grid = (T/64, B*H); 4 waves x 16 q-rows each. q carries gamma_q/sqrt(Dh).
__global__ __launch_bounds__(128) void flash_attn(const _Float16* __restrict__ q,
                                                  const _Float16* __restrict__ k,
                                                  const _Float16* __restrict__ vt,
                                                  _Float16* __restrict__ ctx) {
  __shared__ __align__(16) _Float16 pTile[4][16 * 32];  // per-wave P (16x32)

  const int lane = threadIdx.x & 31, wave = threadIdx.x >> 5;
  const int ln = lane & 15, hf = lane >> 4;
  const int bh = blockIdx.y, b = bh >> 4, h = bh & 15;
  const int qRow0 = blockIdx.x * 64 + wave * 16;

  const _Float16* qBase = q + (size_t)b * T_SEQ * D_MODEL + h * D_HEAD;
  const _Float16* kBase = k + (size_t)b * T_SEQ * D_MODEL + h * D_HEAD;
  const _Float16* vBase = vt + (size_t)bh * D_HEAD * T_SEQ;  // [d][t]
  _Float16* pW = &pTile[wave][0];

  // Q tile 16x128 as 4 A-fragments (contraction chunks of 32 over d)
  v16h qF[4];
  {
    const _Float16* qr = qBase + (size_t)(qRow0 + ln) * D_MODEL;
#pragma unroll
    for (int c = 0; c < 4; ++c) qF[c] = load_a16(qr, c * 32, hf);
  }

  v8f oAcc[8] = {};
  float mS[8], lS[8];
#pragma unroll
  for (int r = 0; r < 8; ++r) { mS[r] = -1e30f; lS[r] = 0.f; }

  // Unmasked blocks: ks + 31 <= qRow0  <=>  ks < kClean
  const int kClean = (qRow0 + 1) & ~31;
  for (int ks = 0; ks < kClean; ks += 32)
    flash_step<false>(ks, qRow0, ln, hf, qF, kBase, vBase, pW, oAcc, mS, lS);
  // Exactly one diagonal (masked) block per wave covers keys up to qRow0+15
  flash_step<true>(kClean, qRow0, ln, hf, qF, kBase, vBase, pW, oAcc, mS, lS);

  // ---- normalize and store context (f16, row-major [b*T+t, h*128+d]) ----
#pragma unroll
  for (int r = 0; r < 8; ++r) {
    const float inv = 1.f / lS[r];
    const int row = qRow0 + r + 8 * hf;
    _Float16* dst = ctx + ((size_t)b * T_SEQ + row) * D_MODEL + h * D_HEAD + ln;
#pragma unroll
    for (int nt = 0; nt < 8; ++nt)
      dst[nt * 16] = (_Float16)(oAcc[nt][r] * inv);
  }
}

// ---------------------------------------------------------------------------
// Host-side orchestration
// ---------------------------------------------------------------------------
extern "C" void kernel_launch(void* const* d_in, const int* in_sizes, int n_in,
                              void* d_out, int out_size, void* d_ws,
                              size_t ws_size, hipStream_t stream) {
  const float* x  = (const float*)d_in[0];
  const float* wSrc[4] = {(const float*)d_in[1], (const float*)d_in[2],
                          (const float*)d_in[3], (const float*)d_in[4]};

  char* ws = (char*)d_ws;
  float* gammas = (float*)ws;                    // 4 floats
  float* part   = (float*)(ws + 4096);           // 1024 floats
  const size_t MB = 1024ull * 1024ull;
  char* base = ws + 65536;
  _Float16* wH[4];
  for (int j = 0; j < 4; ++j) wH[j] = (_Float16*)(base + (size_t)j * 8 * MB);
  _Float16* xH  = (_Float16*)(base + 32 * MB);   // 32MB
  _Float16* qH  = (_Float16*)(base + 64 * MB);   // 32MB
  _Float16* kH  = (_Float16*)(base + 96 * MB);   // 32MB
  _Float16* vT  = (_Float16*)(base + 128 * MB);  // 32MB [b,h,d,t]
  _Float16* ctx = xH;                            // reuse x_h after projections

  // gamma + ternary quantization per weight
  for (int j = 0; j < 4; ++j) {
    abs_sum_partial<<<1024, 256, 0, stream>>>((const float4*)wSrc[j], part,
                                              W_ELEMS / 4);
    gamma_final<<<1, 256, 0, stream>>>(part, 1024, 1.f / (float)W_ELEMS,
                                       gammas + j);
    quant_w<<<(W_ELEMS / 4 + 255) / 256, 256, 0, stream>>>(
        (const float4*)wSrc[j], gammas + j, (v4h*)wH[j], W_ELEMS / 4);
  }
  cvt_f16<<<(M_ROWS * D_MODEL / 4 + 255) / 256, 256, 0, stream>>>(
      (const float4*)x, (v4h*)xH, M_ROWS * D_MODEL / 4);

  dim3 gGrid(D_MODEL / 128, M_ROWS / 128);  // (16, 64)
  const float invSqrtD = 0.08838834764831845f;  // 1/sqrt(128)
  gemm_bit<0><<<gGrid, 128, 0, stream>>>(xH, wH[0], gammas + 0, qH, invSqrtD);
  gemm_bit<0><<<gGrid, 128, 0, stream>>>(xH, wH[1], gammas + 1, kH, 1.f);
  gemm_bit<2><<<gGrid, 128, 0, stream>>>(xH, wH[2], gammas + 2, vT, 1.f);

  flash_attn<<<dim3(T_SEQ / 64, B_SZ * N_HEAD), 128, 0, stream>>>(qH, kH, vT,
                                                                  ctx);

  gemm_bit<1><<<gGrid, 128, 0, stream>>>(ctx, wH[3], gammas + 3, d_out, 1.f);
}
